// RCTV2_88158498718223
// MI455X (gfx1250) — compile-verified, matrix operand order checked by reading, not run on hardware
//
#include <hip/hip_runtime.h>

typedef __bf16 bf16_t;
typedef bf16_t v16bf __attribute__((ext_vector_type(16)));
typedef bf16_t v8bf  __attribute__((ext_vector_type(8)));
typedef float  v8f   __attribute__((ext_vector_type(8)));

#define NB 2
#define NC 256
#define NPIX 4096
#define NPAD 4356  // 66*66 padded pixel rows
#define NEGV (-1000000.0f)

static __device__ __forceinline__ v8f zero8() {
  v8f z;
#pragma unroll
  for (int i = 0; i < 8; ++i) z[i] = 0.0f;
  return z;
}

// CDNA5 16-bit A/B frag: lane L holds row (L%16); elements are two contiguous
// 8-wide K runs at offsets 8*half and 16+8*half -> two aligned 16B loads.
static __device__ __forceinline__ v16bf frag_g(const bf16_t* row, int half) {
  union { v16bf f; v8bf h[2]; } u;
  u.h[0] = *(const v8bf*)(row + 8 * half);
  u.h[1] = *(const v8bf*)(row + 16 + 8 * half);
  return u.f;
}

static __device__ __forceinline__ v8f wmma_bf16(v16bf a, v16bf b, v8f c) {
  return __builtin_amdgcn_wmma_f32_16x16x32_bf16(false, a, false, b, (short)0, c, false, false);
}

// ---------------------------------------------------------------- mask pool
__global__ __launch_bounds__(256) void k_pool(const float* __restrict__ mask,
                                              float* __restrict__ M) {
  int idx = blockIdx.x * 256 + threadIdx.x;
  if (idx >= NB * NPIX) return;
  int b = idx >> 12, p = idx & 4095;
  int py = p >> 6, px = p & 63;
  const float* mp = mask + (size_t)b * 512 * 512 + (size_t)(py * 8) * 512 + px * 8;
  float mx = 0.f;
#pragma unroll
  for (int dy = 0; dy < 8; ++dy)
#pragma unroll
    for (int dx = 0; dx < 8; ++dx) mx = fmaxf(mx, mp[dy * 512 + dx]);
  M[idx] = (mx > 0.f) ? 1.f : 0.f;
}

// ---------------------------------------------------------------- casts
__global__ __launch_bounds__(256) void k_cast(const float* __restrict__ s,
                                              bf16_t* __restrict__ d, int n) {
  int i = blockIdx.x * 256 + threadIdx.x;
  if (i < n) d[i] = (bf16_t)s[i];
}

// vb[b][c][n] = background-masked x, bf16
__global__ __launch_bounds__(256) void k_maskcast(const float* __restrict__ x,
                                                  const float* __restrict__ M,
                                                  bf16_t* __restrict__ vb) {
  int idx = blockIdx.x * 256 + threadIdx.x;  // NB*NC*NPIX = 2^21
  int b = idx >> 20, pix = idx & 4095;
  vb[idx] = (M[b * NPIX + pix] > 0.f) ? (bf16_t)0.0f : (bf16_t)x[idx];
}

// padded transpose+cast: [b][c][64][64] f32 -> [b][66*66][c] bf16, halo = 0
__global__ __launch_bounds__(256) void k_padT(const float* __restrict__ src,
                                              bf16_t* __restrict__ dst) {
  const int tid = threadIdx.x, pj = tid & 31, cg = tid >> 5;
  const int bz = blockIdx.x / 137;
  const int row = (blockIdx.x % 137) * 32 + pj;
  const int yy = row / 66, xx = row % 66;
  const int iy = yy - 1, ix = xx - 1;
  const bool valid = ((unsigned)iy < 64u) && ((unsigned)ix < 64u);
  const int pix = valid ? (iy * 64 + ix) : 0;
  const size_t sb = (size_t)bz * NC * NPIX + pix;
  const size_t db = ((size_t)bz * NPAD + row) * NC + cg * 32;
  if (row >= NPAD) return;
#pragma unroll
  for (int j8 = 0; j8 < 32; j8 += 8) {
    v8bf pk;
#pragma unroll
    for (int j = 0; j < 8; ++j) {
      float v = valid ? src[sb + (size_t)(cg * 32 + j8 + j) * NPIX] : 0.f;
      pk[j] = (bf16_t)v;
    }
    *(v8bf*)(dst + db + j8) = pk;
  }
}

// pack conv weight [co][ci][3][3] -> [r][co][ci] bf16
__global__ __launch_bounds__(256) void k_packw(const float* __restrict__ w,
                                               bf16_t* __restrict__ wp) {
  int idx = blockIdx.x * 256 + threadIdx.x;  // 9*256*256 exactly
  int r = idx >> 16;
  int rem = idx & 65535;
  int co = rem >> 8, ci = rem & 255;
  wp[idx] = (bf16_t)w[(size_t)co * 2304 + ci * 9 + r];
}

// --- implicit-GEMM 3x3 conv (SAME) on padded-transposed input, no LDS/branches
// inP bf16 [b][4356][c], wp bf16 [9][co][ci], out f32 [b][c][n]
// block: 64 co x 256 pix; wave: 32 co x 64 pix (2x4 WMMA tiles)
__global__ __launch_bounds__(256) void k_conv3x3(const bf16_t* __restrict__ inP,
                                                 const bf16_t* __restrict__ wp,
                                                 float* __restrict__ out) {
  const int tid = threadIdx.x, lane = tid & 31, wave = tid >> 5;
  const int lr = lane & 15, half = lane >> 4;
  const int wm = wave >> 2, wn = wave & 3;
  const int bz = blockIdx.z, cot = blockIdx.y * 64, pt0 = blockIdx.x * 256;
  const size_t ob = (size_t)bz * NC * NPIX;
  v8f acc[2][4];
#pragma unroll
  for (int i = 0; i < 2; ++i)
#pragma unroll
    for (int j = 0; j < 4; ++j) acc[i][j] = zero8();

  const bf16_t* aRowB[2];
#pragma unroll
  for (int sm = 0; sm < 2; ++sm)
    aRowB[sm] = wp + (size_t)(cot + wm * 32 + sm * 16 + lr) * 256;
  const bf16_t* bRowB[4];
#pragma unroll
  for (int sn = 0; sn < 4; ++sn) {
    int pix = pt0 + wn * 64 + sn * 16 + lr;
    int py = pix >> 6, px = pix & 63;
    bRowB[sn] = inP + ((size_t)bz * NPAD + (py + 1) * 66 + (px + 1)) * NC;
  }

  for (int r = 0; r < 9; ++r) {
    const int dy = r / 3 - 1, dx = r % 3 - 1;
    const ptrdiff_t boff = (ptrdiff_t)(dy * 66 + dx) * NC;
    const size_t aoff = (size_t)r * 65536;
#pragma unroll
    for (int cc = 0; cc < 8; ++cc) {
      v16bf bfr[4];
#pragma unroll
      for (int sn = 0; sn < 4; ++sn)
        bfr[sn] = frag_g(bRowB[sn] + boff + cc * 32, half);
#pragma unroll
      for (int sm = 0; sm < 2; ++sm) {
        v16bf a = frag_g(aRowB[sm] + aoff + cc * 32, half);
#pragma unroll
        for (int sn = 0; sn < 4; ++sn)
          acc[sm][sn] = wmma_bf16(a, bfr[sn], acc[sm][sn]);
      }
    }
  }
#pragma unroll
  for (int sm = 0; sm < 2; ++sm)
#pragma unroll
    for (int sn = 0; sn < 4; ++sn)
#pragma unroll
      for (int rr = 0; rr < 8; ++rr) {
        int co = cot + wm * 32 + sm * 16 + rr + 8 * half;
        int pix = pt0 + wn * 64 + sn * 16 + lr;
        out[ob + (size_t)co * NPIX + pix] = acc[sm][sn][rr];
      }
}

// --------------------------------------------- instance norm (+relu [+res])
__global__ __launch_bounds__(256) void k_inorm(const float* __restrict__ y,
                                               const float* __restrict__ x,
                                               float* __restrict__ outp, int addres) {
  __shared__ float s1[256], s2[256];
  const int tid = threadIdx.x;
  const size_t base = (size_t)blockIdx.x * NPIX;
  float s = 0.f, q = 0.f;
  for (int i = tid; i < NPIX; i += 256) {
    float v = y[base + i];
    s += v; q += v * v;
  }
  s1[tid] = s; s2[tid] = q;
  __syncthreads();
  for (int st = 128; st > 0; st >>= 1) {
    if (tid < st) { s1[tid] += s1[tid + st]; s2[tid] += s2[tid + st]; }
    __syncthreads();
  }
  const float mu = s1[0] * (1.f / NPIX);
  const float var = s2[0] * (1.f / NPIX) - mu * mu;
  const float inv = rsqrtf(var + 1e-5f);
  for (int i = tid; i < NPIX; i += 256) {
    float v = (y[base + i] - mu) * inv;
    if (addres) v += x[base + i];
    outp[base + i] = fmaxf(v, 0.f);
  }
}

// --------- per-pixel channel L2-normalize + transpose -> fn[b][n][c] bf16
__global__ __launch_bounds__(256) void k_pixnorm(const float* __restrict__ feat,
                                                 bf16_t* __restrict__ fn) {
  __shared__ float red[256];
  __shared__ float sc[32];
  const int tid = threadIdx.x, pj = tid & 31, cg = tid >> 5;
  const int bz = blockIdx.x >> 7;
  const int p0 = (blockIdx.x & 127) * 32;
  const size_t fb = (size_t)bz * NC * NPIX;
  float ss = 0.f;
  for (int j = 0; j < 32; ++j) {
    int ci = cg * 32 + j;
    float v = feat[fb + (size_t)ci * NPIX + p0 + pj];
    ss += v * v;
  }
  red[tid] = ss;
  __syncthreads();
  if (cg == 0) {
    float t = 0.f;
    for (int g = 0; g < 8; ++g) t += red[g * 32 + pj];
    sc[pj] = 1.f / (sqrtf(t) + 1e-8f);
  }
  __syncthreads();
  const float s = sc[pj];
  const size_t db = ((size_t)bz * NPIX + p0 + pj) * NC + cg * 32;
#pragma unroll
  for (int j8 = 0; j8 < 32; j8 += 8) {
    v8bf pk;
#pragma unroll
    for (int j = 0; j < 8; ++j)
      pk[j] = (bf16_t)(feat[fb + (size_t)(cg * 32 + j8 + j) * NPIX + p0 + pj] * s);
    *(v8bf*)(fn + db + j8) = pk;
  }
}

// ------------- attention pass 1: den[q] = sum_k exp(S[q,k]-20), barrier-free
__global__ __launch_bounds__(256) void k_den(const bf16_t* __restrict__ fn,
                                             const float* __restrict__ M,
                                             float* __restrict__ den) {
  __shared__ float dpart[2][64];
  const int tid = threadIdx.x, lane = tid & 31, wave = tid >> 5;
  const int lr = lane & 15, half = lane >> 4;
  const int qsub = wave & 3, khalf = wave >> 2;
  const int bz = blockIdx.y;
  const int q0 = blockIdx.x * 64;
  const bf16_t* fnB = fn + (size_t)bz * NPIX * NC;
  const bf16_t* aRow = fnB + (size_t)(q0 + qsub * 16 + lr) * NC;
  float qp[8];
#pragma unroll
  for (int rr = 0; rr < 8; ++rr)
    qp[rr] = (1.f - M[bz * NPIX + q0 + qsub * 16 + rr + 8 * half]) * NEGV;
  float dacc[8];
#pragma unroll
  for (int rr = 0; rr < 8; ++rr) dacc[rr] = 0.f;

  for (int kt = 0; kt < 64; ++kt) {
    const int k0 = kt * 64;
    const bf16_t* bRow[2];
#pragma unroll
    for (int t2 = 0; t2 < 2; ++t2)
      bRow[t2] = fnB + (size_t)(k0 + (khalf * 2 + t2) * 16 + lr) * NC;
    if (kt < 63)
      __builtin_prefetch(fnB + (size_t)(k0 + 64 + khalf * 32 + lr) * NC, 0, 0);
    v8f sa[2];
    sa[0] = zero8(); sa[1] = zero8();
#pragma unroll
    for (int c8 = 0; c8 < 8; ++c8) {
      v16bf a = frag_g(aRow + c8 * 32, half);
#pragma unroll
      for (int t2 = 0; t2 < 2; ++t2)
        sa[t2] = wmma_bf16(a, frag_g(bRow[t2] + c8 * 32, half), sa[t2]);
    }
    float es[8];
#pragma unroll
    for (int rr = 0; rr < 8; ++rr) es[rr] = 0.f;
#pragma unroll
    for (int t2 = 0; t2 < 2; ++t2) {
      int kl = (khalf * 2 + t2) * 16 + lr;
      float kp = M[bz * NPIX + k0 + kl] * NEGV;
#pragma unroll
      for (int rr = 0; rr < 8; ++rr)
        es[rr] += __expf(sa[t2][rr] * 20.f + qp[rr] + kp - 20.f);
    }
#pragma unroll
    for (int rr = 0; rr < 8; ++rr) {  // deterministic butterfly over 16 lanes
      float v = es[rr];
      v += __shfl_xor(v, 1);
      v += __shfl_xor(v, 2);
      v += __shfl_xor(v, 4);
      v += __shfl_xor(v, 8);
      dacc[rr] += v;
    }
  }
  if (lr == 0) {
#pragma unroll
    for (int rr = 0; rr < 8; ++rr)
      dpart[khalf][qsub * 16 + rr + 8 * half] = dacc[rr];
  }
  __syncthreads();
  if (tid < 64) den[bz * NPIX + q0 + tid] = dpart[0][tid] + dpart[1][tid];
}

// ---- pass 2: att tile in LDS, att_fore = V*att^T -> afb bf16 [b][n][c]
__global__ __launch_bounds__(256) void k_pass2(const bf16_t* __restrict__ fn,
                                               const bf16_t* __restrict__ vb,
                                               const float* __restrict__ M,
                                               const float* __restrict__ den,
                                               bf16_t* __restrict__ afb,
                                               float* __restrict__ visp) {
  __shared__ bf16_t attn[64][64];
  __shared__ float vcol[4][64];
  const int tid = threadIdx.x, lane = tid & 31, wave = tid >> 5;
  const int lr = lane & 15, half = lane >> 4;
  const int qsub = wave & 3, khalf = wave >> 2;
  const int bz = blockIdx.y;
  const int q0 = blockIdx.x * 64;
  const bf16_t* fnB = fn + (size_t)bz * NPIX * NC;
  const bf16_t* vbB = vb + (size_t)bz * NC * NPIX;
  const bf16_t* aRow = fnB + (size_t)(q0 + qsub * 16 + lr) * NC;
  const bf16_t* vRow[2];
#pragma unroll
  for (int fc = 0; fc < 2; ++fc)
    vRow[fc] = vbB + (size_t)((wave * 2 + fc) * 16 + lr) * NPIX;
  float qp[8], qiv[8];
#pragma unroll
  for (int rr = 0; rr < 8; ++rr) {
    int qg = q0 + qsub * 16 + rr + 8 * half;
    qp[rr] = (1.f - M[bz * NPIX + qg]) * NEGV;
    float d = den[bz * NPIX + qg];
    qiv[rr] = (d > 0.f) ? (1.f / d) : 0.f;
  }
  v8f accF[2][4];
#pragma unroll
  for (int fc = 0; fc < 2; ++fc)
#pragma unroll
    for (int qs = 0; qs < 4; ++qs) accF[fc][qs] = zero8();

  for (int kt = 0; kt < 64; ++kt) {
    const int k0 = kt * 64;
    const bf16_t* bRow[2];
#pragma unroll
    for (int t2 = 0; t2 < 2; ++t2)
      bRow[t2] = fnB + (size_t)(k0 + (khalf * 2 + t2) * 16 + lr) * NC;
    if (kt < 63)
      __builtin_prefetch(fnB + (size_t)(k0 + 64 + khalf * 32 + lr) * NC, 0, 0);
    v8f sa[2];
    sa[0] = zero8(); sa[1] = zero8();
#pragma unroll
    for (int c8 = 0; c8 < 8; ++c8) {
      v16bf a = frag_g(aRow + c8 * 32, half);
#pragma unroll
      for (int t2 = 0; t2 < 2; ++t2)
        sa[t2] = wmma_bf16(a, frag_g(bRow[t2] + c8 * 32, half), sa[t2]);
    }
#pragma unroll
    for (int t2 = 0; t2 < 2; ++t2) {
      int kl = (khalf * 2 + t2) * 16 + lr;
      float kp = M[bz * NPIX + k0 + kl] * NEGV;
      float csum = 0.f;
#pragma unroll
      for (int rr = 0; rr < 8; ++rr) {
        int ql = qsub * 16 + rr + 8 * half;
        float e = __expf(sa[t2][rr] * 20.f + qp[rr] + kp - 20.f) * qiv[rr];
        attn[ql][kl] = (bf16_t)e;
        csum += e;
      }
      csum += __shfl_xor(csum, 16);  // combine the two q-row halves
      if (half == 0) vcol[qsub][kl] = csum;
    }
    __syncthreads();
    if (tid < 64)
      visp[((size_t)(bz * 64 + blockIdx.x)) * NPIX + k0 + tid] =
          vcol[0][tid] + vcol[1][tid] + vcol[2][tid] + vcol[3][tid];
#pragma unroll
    for (int kh = 0; kh < 2; ++kh) {
#pragma unroll
      for (int fc = 0; fc < 2; ++fc) {
        v16bf a = frag_g(vRow[fc] + k0 + kh * 32, half);
#pragma unroll
        for (int qs = 0; qs < 4; ++qs)
          accF[fc][qs] =
              wmma_bf16(a, frag_g(&attn[qs * 16 + lr][kh * 32], half), accF[fc][qs]);
      }
    }
    __syncthreads();
  }
  // write att_fore directly as bf16, transposed: afb[b][q][c]
#pragma unroll
  for (int fc = 0; fc < 2; ++fc)
#pragma unroll
    for (int qs = 0; qs < 4; ++qs) {
      int qg = q0 + qs * 16 + lr;
      bf16_t* dst = afb + ((size_t)bz * NPIX + qg) * NC + (wave * 2 + fc) * 16 + 8 * half;
      v8bf pk;
#pragma unroll
      for (int rr = 0; rr < 8; ++rr) pk[rr] = (bf16_t)accF[fc][qs][rr];
      *(v8bf*)dst = pk;
    }
}

// ------------- 1x1 fuse conv over [att_fore^T ; x^T] + final mask compose
// block: 64 co x 256 pix; wave: 32 co x 64 pix (2x4 WMMA tiles)
__global__ __launch_bounds__(256) void k_fuse(const bf16_t* __restrict__ wfb,
                                              const bf16_t* __restrict__ afb,
                                              const bf16_t* __restrict__ xP,
                                              const float* __restrict__ M,
                                              const float* __restrict__ x,
                                              float* __restrict__ out) {
  const int tid = threadIdx.x, lane = tid & 31, wave = tid >> 5;
  const int lr = lane & 15, half = lane >> 4;
  const int wm = wave >> 2, wn = wave & 3;
  const int bz = blockIdx.z, cot = blockIdx.y * 64, pt0 = blockIdx.x * 256;
  const size_t cb = (size_t)bz * NC * NPIX;
  v8f acc[2][4];
#pragma unroll
  for (int i = 0; i < 2; ++i)
#pragma unroll
    for (int j = 0; j < 4; ++j) acc[i][j] = zero8();
  const bf16_t* aRow[2];
  const bf16_t* bRowA[4];
  const bf16_t* bRowX[4];
#pragma unroll
  for (int sm = 0; sm < 2; ++sm)
    aRow[sm] = wfb + (size_t)(cot + wm * 32 + sm * 16 + lr) * 512;
#pragma unroll
  for (int sn = 0; sn < 4; ++sn) {
    int pix = pt0 + wn * 64 + sn * 16 + lr;
    int py = pix >> 6, px = pix & 63;
    bRowA[sn] = afb + ((size_t)bz * NPIX + pix) * NC;
    bRowX[sn] = xP + ((size_t)bz * NPAD + (py + 1) * 66 + (px + 1)) * NC;
  }
#pragma unroll
  for (int cc = 0; cc < 16; ++cc) {
    v16bf bfr[4];
#pragma unroll
    for (int sn = 0; sn < 4; ++sn) {
      const bf16_t* src = (cc < 8) ? (bRowA[sn] + cc * 32) : (bRowX[sn] + (cc - 8) * 32);
      bfr[sn] = frag_g(src, half);
    }
#pragma unroll
    for (int sm = 0; sm < 2; ++sm) {
      v16bf a = frag_g(aRow[sm] + cc * 32, half);
#pragma unroll
      for (int sn = 0; sn < 4; ++sn) acc[sm][sn] = wmma_bf16(a, bfr[sn], acc[sm][sn]);
    }
  }
#pragma unroll
  for (int sm = 0; sm < 2; ++sm)
#pragma unroll
    for (int sn = 0; sn < 4; ++sn)
#pragma unroll
      for (int rr = 0; rr < 8; ++rr) {
        int co = cot + wm * 32 + sm * 16 + rr + 8 * half;
        int pix = pt0 + wn * 64 + sn * 16 + lr;
        size_t oi = cb + (size_t)co * NPIX + pix;
        out[oi] = (M[bz * NPIX + pix] > 0.f) ? acc[sm][sn][rr] : x[oi];
      }
}

// --------------------------------------------- visatt reduce / max / upsample
__global__ __launch_bounds__(256) void k_visred(const float* __restrict__ visp,
                                                float* __restrict__ vis) {
  int idx = blockIdx.x * 256 + threadIdx.x;
  if (idx >= NB * NPIX) return;
  int b = idx >> 12, k = idx & 4095;
  float s = 0.f;
  for (int qt = 0; qt < 64; ++qt) s += visp[((size_t)(b * 64 + qt)) * NPIX + k];
  vis[idx] = s;
}

__global__ __launch_bounds__(256) void k_maxred(const float* __restrict__ vis,
                                                float* __restrict__ mx) {
  __shared__ float s[256];
  int tid = threadIdx.x;
  float m = 0.f;
  for (int i = tid; i < NB * NPIX; i += 256) m = fmaxf(m, vis[i]);
  s[tid] = m;
  __syncthreads();
  for (int st = 128; st > 0; st >>= 1) {
    if (tid < st) s[tid] = fmaxf(s[tid], s[tid + st]);
    __syncthreads();
  }
  if (tid == 0) mx[0] = s[0];
}

__global__ __launch_bounds__(256) void k_upsample(const float* __restrict__ vis,
                                                  const float* __restrict__ mx,
                                                  float* __restrict__ out2) {
  int idx = blockIdx.x * 256 + threadIdx.x;
  if (idx >= NB * 512 * 512) return;
  int b = idx / (512 * 512);
  int rem = idx % (512 * 512);
  int y = rem >> 9, xx = rem & 511;
  out2[idx] = vis[b * NPIX + (y >> 3) * 64 + (xx >> 3)] / mx[0];
}

// ---------------------------------------------------------------- launcher
extern "C" void kernel_launch(void* const* d_in, const int* in_sizes, int n_in,
                              void* d_out, int out_size, void* d_ws, size_t ws_size,
                              hipStream_t stream) {
  (void)in_sizes; (void)n_in; (void)out_size; (void)ws_size;
  const float* x    = (const float*)d_in[0];
  const float* mask = (const float*)d_in[1];
  const float* w1   = (const float*)d_in[2];
  const float* w2   = (const float*)d_in[3];
  const float* wf   = (const float*)d_in[4];
  float* out  = (float*)d_out;
  float* out2 = out + (size_t)NB * NC * NPIX;

  char* ws = (char*)d_ws;
  size_t off = 0;
  auto alloc = [&](size_t bytes) -> void* {
    void* p = ws + off;
    off += (bytes + 255) & ~(size_t)255;
    return p;
  };
  float*  M    = (float*)alloc((size_t)NB * NPIX * 4);
  bf16_t* xP   = (bf16_t*)alloc((size_t)NB * NPAD * NC * 2);
  bf16_t* h1P  = (bf16_t*)alloc((size_t)NB * NPAD * NC * 2);
  bf16_t* wp1  = (bf16_t*)alloc((size_t)9 * 256 * 256 * 2);
  bf16_t* wp2  = (bf16_t*)alloc((size_t)9 * 256 * 256 * 2);
  bf16_t* wfb  = (bf16_t*)alloc((size_t)256 * 512 * 2);
  bf16_t* vb   = (bf16_t*)alloc((size_t)NB * NC * NPIX * 2);
  float*  ybuf = (float*)alloc((size_t)NB * NC * NPIX * 4);
  float*  nbuf = (float*)alloc((size_t)NB * NC * NPIX * 4);
  bf16_t* fn   = (bf16_t*)alloc((size_t)NB * NPIX * NC * 2);
  float*  den  = (float*)alloc((size_t)NB * NPIX * 4);
  bf16_t* afb  = (bf16_t*)alloc((size_t)NB * NPIX * NC * 2);
  float*  visp = (float*)alloc((size_t)NB * 64 * NPIX * 4);
  float*  vis  = (float*)alloc((size_t)NB * NPIX * 4);
  float*  mx   = (float*)alloc(256);

  k_pool<<<(NB * NPIX + 255) / 256, 256, 0, stream>>>(mask, M);
  k_padT<<<NB * 137, 256, 0, stream>>>(x, xP);
  k_maskcast<<<(NB * NC * NPIX) / 256, 256, 0, stream>>>(x, M, vb);
  k_packw<<<(9 * 256 * 256) / 256, 256, 0, stream>>>(w1, wp1);
  k_packw<<<(9 * 256 * 256) / 256, 256, 0, stream>>>(w2, wp2);
  k_cast<<<(256 * 512) / 256, 256, 0, stream>>>(wf, wfb, 256 * 512);

  k_conv3x3<<<dim3(16, 4, NB), 256, 0, stream>>>(xP, wp1, ybuf);
  k_inorm<<<NB * NC, 256, 0, stream>>>(ybuf, nullptr, nbuf, 0);
  k_padT<<<NB * 137, 256, 0, stream>>>(nbuf, h1P);
  k_conv3x3<<<dim3(16, 4, NB), 256, 0, stream>>>(h1P, wp2, ybuf);
  k_inorm<<<NB * NC, 256, 0, stream>>>(ybuf, x, nbuf, 1);

  k_pixnorm<<<NB * 128, 256, 0, stream>>>(nbuf, fn);
  k_den<<<dim3(64, NB), 256, 0, stream>>>(fn, M, den);
  k_pass2<<<dim3(64, NB), 256, 0, stream>>>(fn, vb, M, den, afb, visp);

  k_fuse<<<dim3(16, 4, NB), 256, 0, stream>>>(wfb, afb, xP, M, x, out);

  k_visred<<<(NB * NPIX + 255) / 256, 256, 0, stream>>>(visp, vis);
  k_maxred<<<1, 256, 0, stream>>>(vis, mx);
  k_upsample<<<(NB * 512 * 512) / 256, 256, 0, stream>>>(vis, mx, out2);
}
